// AlternatingForecastModel_70214125355240
// MI455X (gfx1250) — matrix operand order repaired and checked by reading, not compiled, not run on hardware
//
#include <hip/hip_runtime.h>
#include <hip/hip_bf16.h>

typedef __attribute__((ext_vector_type(16))) __bf16 v16bf;
typedef __attribute__((ext_vector_type(8)))  __bf16 v8bf;
typedef __attribute__((ext_vector_type(8)))  float  v8f;

#define T_STEPS 4096
#define BATCH   64
#define HID     512
#define G4      2048      // 4*HID
#define FEAT    16
#define INP_K   32        // input width padded 18 -> 32 for one WMMA K-step
#define WINDOW  168
#define WARMUP  1344
#define WEEK    672
#define MIN_REQ 84
#define MAD_KC  1.4826f
#define BIGV    1e30f
#define BUFSZ   2816      // residual writes happen only for t in [WARMUP, T): <= 2752

#define NT      512       // 16 waves -> 4 waves/SIMD -> ~256 VGPRs/wave budget
#define BLKS_PER_WAVE 8   // 128 (btile,jtile) blocks / 16 waves

// ---------------- prep kernels (weights -> bf16, bias combine) ----------------

__global__ void prep_whh_bf16(const float* __restrict__ w, __bf16* __restrict__ o, int n) {
    int i = blockIdx.x * blockDim.x + threadIdx.x;
    if (i < n) o[i] = (__bf16)w[i];
}

__global__ void prep_wih_bf16(const float* __restrict__ w, __bf16* __restrict__ o) {
    int i = blockIdx.x * blockDim.x + threadIdx.x;   // over G4*INP_K
    if (i < G4 * INP_K) {
        int j = i >> 5, k = i & 31;
        o[i] = (__bf16)((k < FEAT + 2) ? w[j * (FEAT + 2) + k] : 0.0f);
    }
}

__global__ void prep_bias(const float* __restrict__ bi, const float* __restrict__ bh,
                          float* __restrict__ o) {
    int i = blockIdx.x * blockDim.x + threadIdx.x;
    if (i < G4) o[i] = bi[i] + bh[i];
}

// ---------------- persistent LSTM kernel (1 workgroup, 16 waves) ----------------

__global__ __launch_bounds__(NT, 1)
void lstm_forecast_gfx1250(const float* __restrict__ x,
                           const __bf16* __restrict__ Whh,   // [2048][512] bf16
                           const __bf16* __restrict__ Wih,   // [2048][32]  bf16 (padded)
                           const float*  __restrict__ biasc, // [2048] b_ih+b_hh
                           const float*  __restrict__ Wout,  // [512]
                           const float*  __restrict__ bout,  // [1]
                           float* __restrict__ out)          // outputs | h | c
{
    __shared__ __bf16 sh_h[2][BATCH][HID];    // 128 KB, double-buffered hidden state
    __shared__ float  sh_c[BATCH][HID];       // 128 KB, cell state (per-block exclusive)
    __shared__ __bf16 sh_inp[BATCH][INP_K];   // 4 KB padded input tile
    __shared__ float  sh_buf[BUFSZ];          // residual history
    __shared__ float  sh_pred[BATCH];
    __shared__ float  sh_wout[HID];
    __shared__ float  sh_win[WINDOW];
    __shared__ float  sh_wab[WINDOW];
    __shared__ float  sh_red[BATCH][4];
    __shared__ float  sh_med, sh_mad;
    __shared__ int    sh_cnt, sh_m, sh_start, sh_dz, sh_usepred, sh_anom;

    const int tid  = threadIdx.x;
    const int lane = tid & 31;
    const int wv   = tid >> 5;      // wave id 0..15
    const int half = lane >> 4;     // 0/1
    const int nco  = lane & 15;     // column / row-in-tile index

    // ---- init state ----
    for (int i = tid; i < BATCH * HID; i += NT) {
        ((__bf16*)sh_h)[i] = (__bf16)0.0f;
        ((float*)sh_c)[i]  = 0.0f;
    }
    for (int i = tid; i < HID; i += NT) sh_wout[i] = Wout[i];
    if (tid < BATCH) sh_pred[tid] = 0.0f;
    if (tid == 0) sh_cnt = 0;
    __syncthreads();

    const float bo = bout[0];
    int rd = 0;

    for (int t = 0; t < T_STEPS; ++t) {
        // ---------- anomaly detection (scalar path, batch element 0) ----------
        if (tid == 0) {
            float residual0 = x[(size_t)t * FEAT] - sh_pred[0];
            int do_z = (t >= WARMUP) && (t > 0);
            int cnt = sh_cnt;
            if (do_z) sh_buf[cnt] = residual0;
            int n_valid = cnt + do_z;
            int m = n_valid < WINDOW ? n_valid : WINDOW;
            int start = n_valid - WINDOW; if (start < 0) start = 0;
            sh_m = m; sh_start = start; sh_dz = do_z;
            sh_med = 0.0f; sh_mad = 0.0f;
        }
        __syncthreads();
        if (tid < WINDOW) sh_win[tid] = (tid < sh_m) ? sh_buf[sh_start + tid] : BIGV;
        __syncthreads();
        if (tid < WINDOW && sh_m > 0) {            // lower median via rank selection
            float v = sh_win[tid];
            int c = 0;
            for (int j = 0; j < WINDOW; ++j) {
                float w = sh_win[j];
                c += (w < v) || (w == v && j < tid);
            }
            if (c == ((sh_m - 1) >> 1)) sh_med = v;
        }
        __syncthreads();
        if (tid < WINDOW) sh_wab[tid] = (tid < sh_m) ? fabsf(sh_win[tid] - sh_med) : BIGV;
        __syncthreads();
        if (tid < WINDOW && sh_m > 0) {            // MAD via rank selection
            float v = sh_wab[tid];
            int c = 0;
            for (int j = 0; j < WINDOW; ++j) {
                float w = sh_wab[j];
                c += (w < v) || (w == v && j < tid);
            }
            if (c == ((sh_m - 1) >> 1)) sh_mad = v;
        }
        __syncthreads();
        if (tid == 0) {
            float residual0 = x[(size_t)t * FEAT] - sh_pred[0];
            int in_warm = (t < WARMUP);
            int do_z = sh_dz;
            int n_valid = sh_cnt + do_z;
            int valid_z = do_z && (n_valid >= MIN_REQ) && (sh_mad != 0.0f);
            float z = valid_z ? (residual0 - sh_med) / (sh_mad * MAD_KC) : 0.0f;
            int is_anom = valid_z && (fabsf(z) > 3.0f);
            int wm = (in_warm && ((t / WEEK) & 1)) ? 1 : 0;
            int use_orig = in_warm ? (wm == 0) : (!is_anom);
            if (t == 0) use_orig = 1;
            sh_usepred = (!use_orig) && (t > 0);
            sh_anom = in_warm ? 0 : is_anom;
            sh_cnt = n_valid;
        }
        __syncthreads();

        // ---------- build padded bf16 input tile [64][32] ----------
        {
            int use_pred = sh_usepred, anom = sh_anom;
            for (int i = tid; i < BATCH * INP_K; i += NT) {
                int b = i >> 5, k = i & 31;
                float v;
                if (k == 0)             v = use_pred ? sh_pred[b] : x[((size_t)b * T_STEPS + t) * FEAT];
                else if (k < FEAT)      v = x[((size_t)b * T_STEPS + t) * FEAT + k];
                else if (k == FEAT)     v = use_pred ? 1.0f : 0.0f;
                else if (k == FEAT + 1) v = (float)anom;
                else                    v = 0.0f;
                sh_inp[b][k] = (__bf16)v;
            }
        }
        __syncthreads();

        // ---------- WMMA GEMM + fused LSTM cell ----------
        {
            const int wr = rd ^ 1;
#pragma unroll 1                              // one block in flight: 32 live acc VGPRs
            for (int it = 0; it < BLKS_PER_WAVE; ++it) {
                int blk   = wv * BLKS_PER_WAVE + it;  // 128 blocks over 16 waves
                int btile = blk >> 5;                 // 0..3   (batch tiles of 16)
                int jtile = blk & 31;                 // 0..31  (hidden tiles of 16)
                int jg    = jtile * 16 + nco;         // hidden unit for this lane column

                v8f acc[4];
#pragma unroll
                for (int q = 0; q < 4; ++q) {
                    float bsum = biasc[q * HID + jg];
#pragma unroll
                    for (int e = 0; e < 8; ++e) acc[q][e] = bsum;
                }

                // input contribution: one K=32 WMMA per gate
                {
                    const __bf16* arow = &sh_inp[btile * 16 + nco][0];
                    v8bf lo = *(const v8bf*)(arow + half * 16);
                    v8bf hi = *(const v8bf*)(arow + half * 16 + 8);
                    v16bf a = __builtin_shufflevector(lo, hi,
                        0,1,2,3,4,5,6,7,8,9,10,11,12,13,14,15);
#pragma unroll
                    for (int q = 0; q < 4; ++q) {
                        v16bf bm = *(const v16bf*)(Wih + (size_t)(q * HID + jg) * INP_K + half * 16);
                        acc[q] = __builtin_amdgcn_wmma_f32_16x16x32_bf16(
                            false, a, false, bm, (short)0, acc[q], false, false);
                    }
                }
                // hidden recurrence: K = 512 in 16 WMMA steps per gate
#pragma unroll 4
                for (int k0 = 0; k0 < HID; k0 += 32) {
                    const __bf16* arow = &sh_h[rd][btile * 16 + nco][k0];
                    v8bf lo = *(const v8bf*)(arow + half * 16);
                    v8bf hi = *(const v8bf*)(arow + half * 16 + 8);
                    v16bf a = __builtin_shufflevector(lo, hi,
                        0,1,2,3,4,5,6,7,8,9,10,11,12,13,14,15);
#pragma unroll
                    for (int q = 0; q < 4; ++q) {
                        v16bf bm = *(const v16bf*)(Whh + (size_t)(q * HID + jg) * HID + k0 + half * 16);
                        acc[q] = __builtin_amdgcn_wmma_f32_16x16x32_bf16(
                            false, a, false, bm, (short)0, acc[q], false, false);
                    }
                }
                // fused LSTM epilogue (torch gate order i,f,g,o); c lives in LDS,
                // each (btile,jtile) block is touched by exactly one wave -> no races
#pragma unroll
                for (int e = 0; e < 8; ++e) {
                    float si = 1.0f / (1.0f + __expf(-acc[0][e]));
                    float sf = 1.0f / (1.0f + __expf(-acc[1][e]));
                    float tg = tanhf(acc[2][e]);
                    float so = 1.0f / (1.0f + __expf(-acc[3][e]));
                    int brow = btile * 16 + e + half * 8;     // C/D layout: M = v + 8*half
                    float cn = sf * sh_c[brow][jg] + si * tg;
                    sh_c[brow][jg] = cn;
                    float hn = so * tanhf(cn);
                    sh_h[wr][brow][jg] = (__bf16)hn;
                }
            }
        }
        __syncthreads();

        // ---------- output projection: out = h_new @ W_out^T + b_out ----------
        if (tid < 256) {
            int b = tid >> 2, part = tid & 3;
            const __bf16* hrow = &sh_h[rd ^ 1][b][part * 128];
            float s = 0.0f;
            for (int j = 0; j < 128; ++j) s += (float)hrow[j] * sh_wout[part * 128 + j];
            sh_red[b][part] = s;
        }
        __syncthreads();
        if (tid < BATCH) {
            float o = sh_red[tid][0] + sh_red[tid][1] + sh_red[tid][2] + sh_red[tid][3] + bo;
            sh_pred[tid] = o;
            out[(size_t)tid * T_STEPS + t] = o;    // outputs (B, T, 1)
        }
        __syncthreads();
        rd ^= 1;
    }

    // ---------- final (h, c) state dump ----------
    {
        float* hout = out + (size_t)BATCH * T_STEPS;
        float* cout = hout + (size_t)BATCH * HID;
        for (int i = tid; i < BATCH * HID; i += NT) {
            int b = i >> 9, j = i & 511;
            hout[i] = (float)sh_h[rd][b][j];
            cout[i] = ((const float*)sh_c)[i];
        }
    }
}

// ---------------- launch ----------------

extern "C" void kernel_launch(void* const* d_in, const int* in_sizes, int n_in,
                              void* d_out, int out_size, void* d_ws, size_t ws_size,
                              hipStream_t stream) {
    const float* x    = (const float*)d_in[0];
    const float* Wih  = (const float*)d_in[1];
    const float* Whh  = (const float*)d_in[2];
    const float* bih  = (const float*)d_in[3];
    const float* bhh  = (const float*)d_in[4];
    const float* Wout = (const float*)d_in[5];
    const float* bout = (const float*)d_in[6];
    float* out = (float*)d_out;

    // workspace layout: Whh bf16 (2 MB) | Wih bf16 padded (128 KB) | bias f32 (8 KB)
    __bf16* whh_bf = (__bf16*)d_ws;
    __bf16* wih_bf = whh_bf + (size_t)G4 * HID;
    float*  biasc  = (float*)(wih_bf + (size_t)G4 * INP_K);

    prep_whh_bf16<<<(G4 * HID + 255) / 256, 256, 0, stream>>>(Whh, whh_bf, G4 * HID);
    prep_wih_bf16<<<(G4 * INP_K + 255) / 256, 256, 0, stream>>>(Wih, wih_bf);
    prep_bias<<<(G4 + 255) / 256, 256, 0, stream>>>(bih, bhh, biasc);

    lstm_forecast_gfx1250<<<1, NT, 0, stream>>>(x, whh_bf, wih_bf, biasc, Wout, bout, out);
}
